// MemoryMultiAttention_76261439307942
// MI455X (gfx1250) — matrix operand order repaired and checked by reading, not compiled, not run on hardware
//
#include <hip/hip_runtime.h>
#include <stdint.h>
#include <stddef.h>

// Problem constants (fixed by the reference)
#define TOK    (16*24*325)     // 124800 tokens
#define DMODEL 64
#define HEADS  4
#define DH     16
#define MSLOT  64
#define TILES  (TOK/16)        // 7800 16-token tiles
#define WPB    4               // waves per block
#define NBLOCK (TILES/WPB)     // 1950

typedef _Float16 v16h __attribute__((ext_vector_type(16)));
typedef float    v8f  __attribute__((ext_vector_type(8)));
typedef uint32_t v4u  __attribute__((ext_vector_type(4)));
typedef float    v4f  __attribute__((ext_vector_type(4)));

__device__ __forceinline__ uint32_t pack_h2(float a, float b) {
    union { _Float16 h[2]; uint32_t u; } r;
    r.h[0] = (_Float16)a; r.h[1] = (_Float16)b;
    return r.u;
}

__device__ __forceinline__ v16h load_b_operand(const uint32_t* p) {
    // 8 consecutive u32 per lane = one WMMA B operand (16 f16 = 8 VGPRs)
    union { v16h v; v4u q[2]; } u;
    const v4u* p4 = (const v4u*)p;
    u.q[0] = p4[0]; u.q[1] = p4[1];
    return u.v;
}

__device__ __forceinline__ v8f wmma16(v16h a, v16h b, v8f c) {
    // D = A(16x32 f16) * B(32x16 f16) + C(16x16 f32)
    return __builtin_amdgcn_wmma_f32_16x16x32_f16(
        /*neg_a=*/false, a, /*neg_b=*/false, b,
        /*c_mod=*/(short)0, c, /*reuse_a=*/false, /*reuse_b=*/false);
}

// ---------------------------------------------------------------------------
// Prep kernel (1 block, 256 threads): computes the launch-constant operands.
//   K = mem@Wk+bk, V = mem@Wv+bv          (64x64 each, in LDS)
//   P_h = (Wq_h · K_hᵀ)/8  packed f16 in per-lane WMMA B-layout  (32 KB)
//   Vpack = V_h in per-lane WMMA B-layout                        ( 8 KB)
//   sb[h][m] = (bq_h · K[m,h,:])/8                               ( 1 KB)
// B-layout for a [32k x 16n] f16 block, flat index = lane*8 + j:
//   lanes 0..15 : n=lane,    u32 j = (B[k][n], B[k+1][n]), k = 2j + (j<4 ? 0 : 8)
//   lanes 16..31: n=lane-16, u32 j = (B[k][n], B[k+1][n]), k = 8 + 2j + (j<4 ? 0 : 8)
// ---------------------------------------------------------------------------
__global__ __launch_bounds__(256)
void mma_prep(const float* __restrict__ mem,
              const float* __restrict__ Wq, const float* __restrict__ bq,
              const float* __restrict__ Wk, const float* __restrict__ bk,
              const float* __restrict__ Wv, const float* __restrict__ bv,
              uint32_t* __restrict__ Ppack, uint32_t* __restrict__ Vpack,
              float* __restrict__ sbias)
{
    __shared__ float Kls[64*64];
    __shared__ float Vls[64*64];
    const int tid = threadIdx.x;

    for (int i = tid; i < 64*64; i += 256) {
        int m = i >> 6, c = i & 63;
        float ak = bk[c], av = bv[c];
        for (int k = 0; k < 64; ++k) {
            float mv = mem[m*64 + k];
            ak += mv * Wk[k*64 + c];
            av += mv * Wv[k*64 + c];
        }
        Kls[i] = ak; Vls[i] = av;
    }
    __syncthreads();

    // scaled score bias: sb[h*64+m] = (1/8) * dot(bq_h, K[m,h,:])
    {
        int h = tid >> 6, m = tid & 63;
        float a = 0.f;
        for (int d = 0; d < DH; ++d) a += bq[h*16 + d] * Kls[m*64 + h*16 + d];
        sbias[tid] = a * 0.125f;
    }

    // P packed: flat = ((h*4+t)*2 + c)*256 + lane*8 + j   (8192 u32)
    for (int idx = tid; idx < 8192; idx += 256) {
        int j = idx & 7, lane = (idx >> 3) & 31, c = (idx >> 8) & 1,
            t = (idx >> 9) & 3, h = idx >> 11;
        int n  = lane & 15, m = t*16 + n;
        int kk = c*32 + ((lane < 16) ? 0 : 8) + 2*j + ((j < 4) ? 0 : 8);
        float v0 = 0.f, v1 = 0.f;
        for (int d = 0; d < DH; ++d) {
            float kmhd = Kls[m*64 + h*16 + d];
            v0 += Wq[kk*64     + h*16 + d] * kmhd;
            v1 += Wq[(kk+1)*64 + h*16 + d] * kmhd;
        }
        Ppack[idx] = pack_h2(v0 * 0.125f, v1 * 0.125f);
    }

    // V packed: flat = (h*2 + c)*256 + lane*8 + j   (2048 u32)
    for (int idx = tid; idx < 2048; idx += 256) {
        int j = idx & 7, lane = (idx >> 3) & 31, c = (idx >> 8) & 1, h = idx >> 9;
        int n  = lane & 15;
        int kk = c*32 + ((lane < 16) ? 0 : 8) + 2*j + ((j < 4) ? 0 : 8);
        Vpack[idx] = pack_h2(Vls[kk*64 + h*16 + n], Vls[(kk+1)*64 + h*16 + n]);
    }
}

// ---------------------------------------------------------------------------
// Main kernel: one wave per 16-token tile.
//   scores_h = x_tile(16x64) @ P_h(64x64) + sb_h   -> 4 C-layout f32 accs
//   softmax over 64 slots (shfl_xor within 16-lane halves)
//   weights -> per-head LDS buffer (f16, row major) -> reload in A-layout
//   read_h  = w(16x64) @ V_h(64x16); out = read + x
// Head loop fully unrolled with disjoint LDS buffers so head h+1's score
// WMMAs can overlap head h's softmax / LDS round-trip.
// ---------------------------------------------------------------------------
__global__ __launch_bounds__(WPB*32)
void mma_attn(const float* __restrict__ x,
              const uint32_t* __restrict__ Ppack,
              const uint32_t* __restrict__ Vpack,
              const float* __restrict__ sbias,
              float* __restrict__ out)
{
    __shared__ _Float16 wbuf[WPB][HEADS][16*64];   // 2 KB per wave per head

    const int lane = threadIdx.x & 31;
    const int wv   = threadIdx.x >> 5;
    const int tile = blockIdx.x * WPB + wv;
    const int rlo  = lane & 15;
    const int hi16 = lane >> 4;          // 0: rows/k-lo half, 1: hi half
    const int kb   = hi16 ? 8 : 0;       // A-layout k base for this half

    // --- load x tile directly into WMMA A-layout (f16), both k-chunks ---
    const float* xr = x + (size_t)(tile*16 + rlo) * DMODEL;
    __builtin_prefetch(xr, 0, 3);        // global_prefetch_b8 (epilogue re-read)
    const v4f* xf = (const v4f*)xr;
    const int b4 = kb >> 2;

    v16h A0, A1;
#define FILL4(Av, o, f) do { Av[(o)]=(_Float16)(f).x; Av[(o)+1]=(_Float16)(f).y; \
                             Av[(o)+2]=(_Float16)(f).z; Av[(o)+3]=(_Float16)(f).w; } while(0)
    { v4f f;
      f = xf[b4+0];  FILL4(A0, 0,  f);
      f = xf[b4+1];  FILL4(A0, 4,  f);
      f = xf[b4+4];  FILL4(A0, 8,  f);
      f = xf[b4+5];  FILL4(A0, 12, f);
      f = xf[b4+8];  FILL4(A1, 0,  f);
      f = xf[b4+9];  FILL4(A1, 4,  f);
      f = xf[b4+12]; FILL4(A1, 8,  f);
      f = xf[b4+13]; FILL4(A1, 12, f);
    }
#undef FILL4

    const v8f vzero = {0.f,0.f,0.f,0.f,0.f,0.f,0.f,0.f};

#pragma unroll
    for (int h = 0; h < HEADS; ++h) {
        _Float16* wl = wbuf[wv][h];

        // ---- scores: 4 m-tiles x 2 k-chunks = 8 WMMAs ----
        v8f acc[4];
#pragma unroll
        for (int t = 0; t < 4; ++t) {
            const uint32_t* pb = Ppack + (size_t)((h*4 + t)*2) * 256 + lane*8;
            v8f a = vzero;
            a = wmma16(A0, load_b_operand(pb),       a);
            a = wmma16(A1, load_b_operand(pb + 256), a);
            acc[t] = a;
        }

        // ---- add score bias (per m-column, lane-dependent) ----
#pragma unroll
        for (int t = 0; t < 4; ++t) {
            float bias = sbias[h*64 + t*16 + rlo];
#pragma unroll
            for (int r = 0; r < 8; ++r) acc[t][r] += bias;
        }

        // ---- row softmax over 64 slots, fused normalize + LDS store ----
#pragma unroll
        for (int r = 0; r < 8; ++r) {
            float mx = fmaxf(fmaxf(acc[0][r], acc[1][r]),
                             fmaxf(acc[2][r], acc[3][r]));
            mx = fmaxf(mx, __shfl_xor(mx, 1, 32));
            mx = fmaxf(mx, __shfl_xor(mx, 2, 32));
            mx = fmaxf(mx, __shfl_xor(mx, 4, 32));
            mx = fmaxf(mx, __shfl_xor(mx, 8, 32));
            float e0 = exp2f((acc[0][r] - mx) * 1.44269504f);
            float e1 = exp2f((acc[1][r] - mx) * 1.44269504f);
            float e2 = exp2f((acc[2][r] - mx) * 1.44269504f);
            float e3 = exp2f((acc[3][r] - mx) * 1.44269504f);
            float s = e0 + e1 + e2 + e3;
            s += __shfl_xor(s, 1, 32);
            s += __shfl_xor(s, 2, 32);
            s += __shfl_xor(s, 4, 32);
            s += __shfl_xor(s, 8, 32);
            float inv = 1.0f / s;
            _Float16* wr = wl + (r + hi16*8)*64 + rlo;
            wr[0]  = (_Float16)(e0 * inv);
            wr[16] = (_Float16)(e1 * inv);
            wr[32] = (_Float16)(e2 * inv);
            wr[48] = (_Float16)(e3 * inv);
        }

        asm volatile("s_wait_dscnt 0" ::: "memory");

        // ---- reload weights in A-layout (two 128-bit LDS loads / chunk) ----
        v16h W0, W1;
        {
            union { v16h v; v4u q[2]; } u0, u1;
            const v4u* p4 = (const v4u*)(wl + rlo*64 + kb);
            u0.q[0] = p4[0]; u0.q[1] = p4[2];   // k = kb..kb+7 , kb+16..kb+23
            u1.q[0] = p4[4]; u1.q[1] = p4[6];   // +32 halves (second k-chunk)
            W0 = u0.v; W1 = u1.v;
        }

        // ---- read = w @ V_h : 2 WMMAs ----
        const uint32_t* vb = Vpack + (size_t)(h*2) * 256 + lane*8;
        v8f r8 = vzero;
        r8 = wmma16(W0, load_b_operand(vb),       r8);
        r8 = wmma16(W1, load_b_operand(vb + 256), r8);

        // ---- epilogue: out = read + x (residual) ----
        const int colg = h*16 + rlo;
#pragma unroll
        for (int r = 0; r < 8; ++r) {
            size_t off = (size_t)(tile*16 + r + hi16*8) * DMODEL + colg;
            out[off] = r8[r] + x[off];
        }
    }
}

// ---------------------------------------------------------------------------
extern "C" void kernel_launch(void* const* d_in, const int* in_sizes, int n_in,
                              void* d_out, int out_size, void* d_ws, size_t ws_size,
                              hipStream_t stream) {
    const float* x   = (const float*)d_in[0];
    const float* mem = (const float*)d_in[1];
    const float* Wq  = (const float*)d_in[2];
    const float* bq  = (const float*)d_in[3];
    const float* Wk  = (const float*)d_in[4];
    const float* bk  = (const float*)d_in[5];
    const float* Wv  = (const float*)d_in[6];
    const float* bv  = (const float*)d_in[7];

    uint32_t* Pp = (uint32_t*)d_ws;          // 8192 u32 = 32 KB
    uint32_t* Vp = Pp + 8192;                // 2048 u32 =  8 KB
    float*    sb = (float*)(Vp + 2048);      //  256 f32 =  1 KB

    mma_prep<<<1, 256, 0, stream>>>(mem, Wq, bq, Wk, bk, Wv, bv, Pp, Vp, sb);
    mma_attn<<<NBLOCK, WPB*32, 0, stream>>>(x, Pp, Vp, sb, (float*)d_out);
}
